// SE3RefinementBlock_32220844654814
// MI455X (gfx1250) — matrix operand order repaired and checked by reading, not compiled, not run on hardware
//
#include <hip/hip_runtime.h>
#include <hip/hip_bf16.h>
#include <math.h>

#define B_  2
#define N_  512
#define H_  256
#define NH_ 8
#define HD_ 32
#define BN_ (B_*N_)
#define FF_ (4*H_)

typedef __bf16 bf16;
typedef __attribute__((ext_vector_type(16))) __bf16 v16bf;
typedef __attribute__((ext_vector_type(8)))  float  v8f;
typedef __attribute__((ext_vector_type(4)))  float  v4f;

__device__ __forceinline__ float silu_f(float x)   { return x / (1.f + __expf(-x)); }
__device__ __forceinline__ float sigmoid_f(float x){ return 1.f / (1.f + __expf(-x)); }

__device__ __forceinline__ v8f wmma_bf(v16bf a, v16bf b, v8f c) {
  // D = A(16x32 bf16) * B(32x16 bf16) + C(16x16 f32)
  return __builtin_amdgcn_wmma_f32_16x16x32_bf16(false, a, false, b, (short)0, c, false, false);
}
// A fragment: per-lane row m=lane&15, k0=(lane>>4)*8; elements = [k0..k0+7] and [k0+16..k0+23]
__device__ __forceinline__ v16bf ldfragA(const bf16* p) {
  v16bf a;
  ((v4f*)&a)[0] = *(const v4f*)(p);
  ((v4f*)&a)[1] = *(const v4f*)(p + 16);
  return a;
}
// B fragment: per-lane column n=lane&15, k0=(lane>>4)*8; elements = [k0..k0+7] and [k0+16..k0+23]
__device__ __forceinline__ v16bf ldfragB(const bf16* p) {
  v16bf b;
  ((v4f*)&b)[0] = *(const v4f*)(p);
  ((v4f*)&b)[1] = *(const v4f*)(p + 16);
  return b;
}

// ---- weight convert + transpose: out[n*K+k] = (bf16) in[k*Nin + n], zero-pad n>=Nin ----
__global__ void cvt_t_kernel(const float* __restrict__ in, bf16* __restrict__ out,
                             int K, int Nin, int Nout) {
  int idx = blockIdx.x * blockDim.x + threadIdx.x;
  if (idx >= K * Nout) return;
  int n = idx / K, k = idx - n * K;
  out[idx] = (n < Nin) ? (bf16)in[(size_t)k * Nin + n] : (bf16)0.f;
}

// ---- LayerNorm over H=256, one block (256 thr) per token; bf16 output ----
__global__ void ln_kernel(const float* __restrict__ x, const float* __restrict__ g,
                          const float* __restrict__ b, bf16* __restrict__ out) {
  int tok = blockIdx.x, t = threadIdx.x;
  float v = x[(size_t)tok * H_ + t];
  __shared__ float s[H_];
  s[t] = v; __syncthreads();
  for (int o = H_/2; o > 0; o >>= 1) { if (t < o) s[t] += s[t + o]; __syncthreads(); }
  float mu = s[0] * (1.f / H_); __syncthreads();
  float d = v - mu;
  s[t] = d * d; __syncthreads();
  for (int o = H_/2; o > 0; o >>= 1) { if (t < o) s[t] += s[t + o]; __syncthreads(); }
  float var = s[0] * (1.f / H_);
  out[(size_t)tok * H_ + t] = (bf16)(d * rsqrtf(var + 1e-5f) * g[t] + b[t]);
}

// ---- generic bf16 WMMA GEMM, register-blocked 16(M) x 64(N) per wave ----
// C = act(A[M,K] @ Wt[Nn,K]^T + bias)*scale (+residual)(*mask)
// One A-fragment feeds 4 WMMAs per K-step (4x A-traffic reuse).
// bmode for bf16 out: 0 plain [M,Nn]; 1 head-major q/k [b,h,tok,hd]; 2 head-transposed v [b,h,hd,tok]
__global__ void gemm_bf16(const bf16* __restrict__ A, const bf16* __restrict__ Wt,
                          const float* __restrict__ bias, const float* __restrict__ residual,
                          const unsigned char* __restrict__ maskp,
                          float* __restrict__ outF, bf16* __restrict__ outB,
                          int K, int Nn, int act, int bmode, float scale) {
  int lane = threadIdx.x;
  int n0 = blockIdx.x * 64, m0 = blockIdx.y * 16;
  int m = lane & 15, hi = lane >> 4;
  const bf16* arow = A  + (size_t)(m0 + m) * K + hi * 8;
  const bf16* brow = Wt + (size_t)(n0 + m) * K + hi * 8;
  size_t bstep = (size_t)16 * K;
  v8f c0 = {}, c1 = {}, c2 = {}, c3 = {};
  for (int kt = 0; kt < K; kt += 32) {
    v16bf a = ldfragA(arow + kt);
    c0 = wmma_bf(a, ldfragB(brow + kt),             c0);
    c1 = wmma_bf(a, ldfragB(brow + bstep     + kt), c1);
    c2 = wmma_bf(a, ldfragB(brow + bstep * 2 + kt), c2);
    c3 = wmma_bf(a, ldfragB(brow + bstep * 3 + kt), c3);
  }
  v8f acc[4] = {c0, c1, c2, c3};
#pragma unroll
  for (int t = 0; t < 4; ++t) {
    int n = n0 + t * 16 + m;
    float bn = bias ? bias[n] : 0.f;
#pragma unroll
    for (int r = 0; r < 8; ++r) {
      int gm = m0 + r + (hi << 3);                 // global token row
      float val = acc[t][r] + bn;
      if (act) val = silu_f(val);
      val *= scale;
      if (residual) val += residual[(size_t)gm * Nn + n];
      if (maskp && !maskp[gm]) val = 0.f;
      if (outF) outF[(size_t)gm * Nn + n] = val;
      if (outB) {
        size_t idx;
        if (bmode == 0) idx = (size_t)gm * Nn + n;
        else {
          int b = gm >> 9, ti = gm & (N_ - 1), hh = n >> 5, hd = n & (HD_ - 1);
          if (bmode == 1) idx = (((size_t)(b * NH_ + hh) * N_ + ti) << 5) + hd;
          else            idx = (((size_t)(b * NH_ + hh) * HD_ + hd) << 9) + ti;
        }
        outB[idx] = (bf16)val;
      }
    }
  }
}

// ---- distance-bias MLP: db[b,i,j,h] = silu(d*w1+b1) @ w2 + b2, WMMA over K=256 ----
__global__ void db_kernel(const float* __restrict__ coords, const float* __restrict__ w1,
                          const float* __restrict__ b1, const bf16* __restrict__ w2t,
                          const float* __restrict__ b2, float* __restrict__ db) {
  int lane = threadIdx.x;
  int p0 = blockIdx.x * 16;            // 16 consecutive pairs, same (b,i), j=j0..j0+15
  int j0 = p0 & (N_ - 1);
  int bi = p0 >> 9;
  int b = bi >> 9, i = bi & (N_ - 1);
  int m = lane & 15, hi = lane >> 4;
  int j = j0 + m;
  const float* ci = coords + ((size_t)b * N_ + i) * 3;
  const float* cj = coords + ((size_t)b * N_ + j) * 3;
  float dx = cj[0] - ci[0], dy = cj[1] - ci[1], dz = cj[2] - ci[2];
  float d = fmaxf(sqrtf(dx * dx + dy * dy + dz * dz), 1e-6f);
  int ka = hi * 8;
  const bf16* brow = w2t + (size_t)m * H_ + hi * 8;
  v8f c = {};
  for (int kt = 0; kt < H_; kt += 32) {
    v16bf a;
#pragma unroll
    for (int e = 0; e < 16; ++e) {
      int k = kt + ka + (e & 7) + ((e >> 3) << 4);
      float t = d * w1[k] + b1[k];
      a[e] = (bf16)(t / (1.f + __expf(-t)));
    }
    c = wmma_bf(a, ldfragB(brow + kt), c);
  }
  if (m < NH_) {
    float bias = b2[m];
#pragma unroll
    for (int r = 0; r < 8; ++r) {
      int M = r + (hi << 3);
      db[((size_t)(p0 + M)) * NH_ + m] = c[r] + bias;
    }
  }
}

// ---- logits[b,h,i,j] = q.k (K=HD=32, one WMMA) + db, masked on j ----
__global__ void attn_logits_kernel(const bf16* __restrict__ qh, const bf16* __restrict__ kh,
                                   const float* __restrict__ db,
                                   const unsigned char* __restrict__ maskp,
                                   float* __restrict__ logits) {
  int lane = threadIdx.x;
  int jt = blockIdx.x, y = blockIdx.y;
  int it = y & 31, bh = y >> 5;
  int b = bh >> 3, h = bh & 7;
  int m = lane & 15, hi = lane >> 4;
  const bf16* qp = qh + ((size_t)bh * N_ + it * 16 + m) * HD_ + hi * 8;
  const bf16* kp = kh + ((size_t)bh * N_ + jt * 16 + m) * HD_ + hi * 8;
  v8f c = {};
  c = wmma_bf(ldfragA(qp), ldfragB(kp), c);
  int j = jt * 16 + m;
  bool keep = maskp[b * N_ + j] != 0;
#pragma unroll
  for (int r = 0; r < 8; ++r) {
    int i = it * 16 + r + (hi << 3);
    float val = c[r] + db[(((size_t)(b * N_ + i)) * N_ + j) * NH_ + h];
    if (!keep) val = -10000.f;
    logits[((size_t)bh * N_ + i) * N_ + j] = val;
  }
}

// ---- softmax over j (512), wave per row; faithful mask_i + renorm; fp32 + bf16 out ----
__global__ void softmax_kernel(float* __restrict__ attn, bf16* __restrict__ attnb,
                               const unsigned char* __restrict__ maskp) {
  int lane = threadIdx.x;
  int row = blockIdx.x;                 // bh*512 + i
  int bh = row >> 9, b = bh >> 3, i = row & (N_ - 1);
  float* p = attn + (size_t)row * N_;
  float v[16]; float mx = -3.0e38f;
#pragma unroll
  for (int t = 0; t < 16; ++t) { v[t] = p[lane + 32 * t]; mx = fmaxf(mx, v[t]); }
  for (int o = 16; o; o >>= 1) mx = fmaxf(mx, __shfl_xor(mx, o, 32));
  float s = 0.f;
#pragma unroll
  for (int t = 0; t < 16; ++t) { v[t] = __expf(v[t] - mx); s += v[t]; }
  for (int o = 16; o; o >>= 1) s += __shfl_xor(s, o, 32);
  float mi = maskp[b * N_ + i] ? 1.f : 0.f;
  float inv = mi / s;
  float s2 = 0.f;
#pragma unroll
  for (int t = 0; t < 16; ++t) { v[t] *= inv; s2 += v[t]; }
  for (int o = 16; o; o >>= 1) s2 += __shfl_xor(s2, o, 32);
  float inv2 = 1.f / fmaxf(s2, 1e-6f);
#pragma unroll
  for (int t = 0; t < 16; ++t) {
    float a = v[t] * inv2;
    p[lane + 32 * t] = a;
    attnb[(size_t)row * N_ + lane + 32 * t] = (bf16)a;
  }
}

// ---- attn_mean[b,i,j] = mean over heads ----
__global__ void attn_mean_kernel(const float* __restrict__ attn, float* __restrict__ am) {
  int idx = blockIdx.x * blockDim.x + threadIdx.x;       // 0 .. B*N*N-1
  int b = idx >> 18, ij = idx & ((1 << 18) - 1);
  size_t base = (size_t)b * NH_ * (N_ * N_) + ij;
  float s = 0.f;
#pragma unroll
  for (int h = 0; h < NH_; ++h) s += attn[base + (size_t)h * (N_ * N_)];
  am[idx] = s * (1.f / NH_);
}

// ---- msg[b,i, h*32+hd] = attn @ v (K=N=512); both hd-tiles per wave (A reused) ----
__global__ void attn_msg_kernel(const bf16* __restrict__ attnb, const bf16* __restrict__ vT,
                                bf16* __restrict__ msgb) {
  int lane = threadIdx.x;
  int y = blockIdx.x;
  int it = y & 31, bh = y >> 5;
  int m = lane & 15, hi = lane >> 4;
  const bf16* arow  = attnb + ((size_t)bh * N_ + it * 16 + m) * N_ + hi * 8;
  const bf16* brow0 = vT + ((size_t)bh * HD_ + m) * N_ + hi * 8;        // hd tile 0
  const bf16* brow1 = brow0 + (size_t)16 * N_;                           // hd tile 1
  v8f c0 = {}, c1 = {};
  for (int jt = 0; jt < N_; jt += 32) {
    v16bf a = ldfragA(arow + jt);
    c0 = wmma_bf(a, ldfragB(brow0 + jt), c0);
    c1 = wmma_bf(a, ldfragB(brow1 + jt), c1);
  }
  int b = bh >> 3, h = bh & 7;
#pragma unroll
  for (int r = 0; r < 8; ++r) {
    int i = it * 16 + r + (hi << 3);
    size_t base = (size_t)(b * N_ + i) * H_ + h * HD_;
    msgb[base + m]      = (bf16)c0[r];
    msgb[base + 16 + m] = (bf16)c1[r];
  }
}

// ---- gate[tok] = sigmoid(g1 . cg_w2 + cg_b2), wave per token ----
__global__ void gate_kernel(const bf16* __restrict__ g1, const float* __restrict__ w2,
                            const float* __restrict__ b2, float* __restrict__ gate) {
  int lane = threadIdx.x, tok = blockIdx.x;
  float s = 0.f;
#pragma unroll
  for (int t = 0; t < 8; ++t) {
    int k = lane + 32 * t;
    s += (float)g1[(size_t)tok * H_ + k] * w2[k];
  }
  for (int o = 16; o; o >>= 1) s += __shfl_xor(s, o, 32);
  if (lane == 0) gate[tok] = sigmoid_f(s + b2[0]);
}

// ---- coords_out = (c_i + STEP*gate*(sum_j am*c_j - sum_j am * c_i)) * mask_i ----
__global__ void coord_kernel(const float* __restrict__ am, const float* __restrict__ coords,
                             const float* __restrict__ gate, const unsigned char* __restrict__ maskp,
                             float* __restrict__ outc) {
  int lane = threadIdx.x, row = blockIdx.x;   // b*512 + i
  int b = row >> 9;
  float s0 = 0.f, sx = 0.f, sy = 0.f, sz = 0.f;
  for (int t = 0; t < 16; ++t) {
    int j = lane + 32 * t;
    float a = am[(size_t)row * N_ + j];
    const float* cj = coords + ((size_t)b * N_ + j) * 3;
    s0 += a; sx += a * cj[0]; sy += a * cj[1]; sz += a * cj[2];
  }
  for (int o = 16; o; o >>= 1) {
    s0 += __shfl_xor(s0, o, 32); sx += __shfl_xor(sx, o, 32);
    sy += __shfl_xor(sy, o, 32); sz += __shfl_xor(sz, o, 32);
  }
  if (lane < 3) {
    const float* ci = coords + (size_t)row * 3;
    float cc = ci[lane];
    float sv = (lane == 0) ? sx : ((lane == 1) ? sy : sz);
    float delta = sv - s0 * cc;
    float mi = maskp[row] ? 1.f : 0.f;
    outc[(size_t)row * 3 + lane] = (cc + 0.25f * gate[row] * delta) * mi;
  }
}

extern "C" void kernel_launch(void* const* d_in, const int* in_sizes, int n_in,
                              void* d_out, int out_size, void* d_ws, size_t ws_size,
                              hipStream_t stream) {
  (void)in_sizes; (void)n_in; (void)out_size; (void)ws_size;
  const float* hidden = (const float*)d_in[0];
  const float* coords = (const float*)d_in[1];
  const unsigned char* mask = (const unsigned char*)d_in[2];
  const float* hn_g = (const float*)d_in[3];
  const float* hn_b = (const float*)d_in[4];
  const float* ffn_g = (const float*)d_in[5];
  const float* ffn_b = (const float*)d_in[6];
  const float* Wq = (const float*)d_in[7];  const float* bq = (const float*)d_in[8];
  const float* Wk = (const float*)d_in[9];  const float* bk = (const float*)d_in[10];
  const float* Wv = (const float*)d_in[11]; const float* bv = (const float*)d_in[12];
  const float* Wo = (const float*)d_in[13]; const float* bo = (const float*)d_in[14];
  const float* db_w1 = (const float*)d_in[15]; const float* db_b1 = (const float*)d_in[16];
  const float* db_w2 = (const float*)d_in[17]; const float* db_b2 = (const float*)d_in[18];
  const float* cg_w1 = (const float*)d_in[19]; const float* cg_b1 = (const float*)d_in[20];
  const float* cg_w2 = (const float*)d_in[21]; const float* cg_b2 = (const float*)d_in[22];
  const float* ff_w1 = (const float*)d_in[23]; const float* ff_b1 = (const float*)d_in[24];
  const float* ff_w2 = (const float*)d_in[25]; const float* ff_b2 = (const float*)d_in[26];

  float* outHidden = (float*)d_out;                // [B,N,H]
  float* outCoords = (float*)d_out + BN_ * H_;     // [B,N,3]

  // workspace carve-up (~53 MB total)
  char* wsb = (char*)d_ws; size_t off = 0;
  auto alloc = [&](size_t bytes) -> void* {
    void* p = wsb + off; off += (bytes + 255) & ~(size_t)255; return p;
  };
  bf16* Wq_t   = (bf16*)alloc((size_t)H_ * H_ * 2);
  bf16* Wk_t   = (bf16*)alloc((size_t)H_ * H_ * 2);
  bf16* Wv_t   = (bf16*)alloc((size_t)H_ * H_ * 2);
  bf16* Wo_t   = (bf16*)alloc((size_t)H_ * H_ * 2);
  bf16* cg1_t  = (bf16*)alloc((size_t)H_ * H_ * 2);
  bf16* ffw1_t = (bf16*)alloc((size_t)H_ * FF_ * 2);
  bf16* ffw2_t = (bf16*)alloc((size_t)H_ * FF_ * 2);
  bf16* dbw2_t = (bf16*)alloc((size_t)16 * H_ * 2);       // padded 16 x 256
  bf16* h_bf   = (bf16*)alloc((size_t)BN_ * H_ * 2);
  bf16* q_h    = (bf16*)alloc((size_t)BN_ * H_ * 2);      // [b,h,tok,hd]
  bf16* k_h    = (bf16*)alloc((size_t)BN_ * H_ * 2);      // [b,h,tok,hd]
  bf16* vT     = (bf16*)alloc((size_t)BN_ * H_ * 2);      // [b,h,hd,tok]
  float* db    = (float*)alloc((size_t)B_ * N_ * N_ * NH_ * 4);   // [b,i,j,h]
  float* attn  = (float*)alloc((size_t)B_ * NH_ * N_ * N_ * 4);   // logits, then attn
  bf16* attnb  = (bf16*)alloc((size_t)B_ * NH_ * N_ * N_ * 2);
  float* am    = (float*)alloc((size_t)B_ * N_ * N_ * 4);
  bf16* msgb   = (bf16*)alloc((size_t)BN_ * H_ * 2);
  float* hid2  = (float*)alloc((size_t)BN_ * H_ * 4);
  bf16* hff    = (bf16*)alloc((size_t)BN_ * H_ * 2);
  bf16* a1     = (bf16*)alloc((size_t)BN_ * FF_ * 2);
  bf16* h3_bf  = (bf16*)alloc((size_t)BN_ * H_ * 2);
  bf16* g1     = (bf16*)alloc((size_t)BN_ * H_ * 2);
  float* gate  = (float*)alloc((size_t)BN_ * 4);

  // 1) convert+transpose weights to bf16 [out,in]
  cvt_t_kernel<<<(H_*H_)/256, 256, 0, stream>>>(Wq, Wq_t, H_, H_, H_);
  cvt_t_kernel<<<(H_*H_)/256, 256, 0, stream>>>(Wk, Wk_t, H_, H_, H_);
  cvt_t_kernel<<<(H_*H_)/256, 256, 0, stream>>>(Wv, Wv_t, H_, H_, H_);
  cvt_t_kernel<<<(H_*H_)/256, 256, 0, stream>>>(Wo, Wo_t, H_, H_, H_);
  cvt_t_kernel<<<(H_*H_)/256, 256, 0, stream>>>(cg_w1, cg1_t, H_, H_, H_);
  cvt_t_kernel<<<(H_*FF_)/256, 256, 0, stream>>>(ff_w1, ffw1_t, H_, FF_, FF_);   // [1024,256]
  cvt_t_kernel<<<(H_*FF_)/256, 256, 0, stream>>>(ff_w2, ffw2_t, FF_, H_, H_);    // [256,1024]
  cvt_t_kernel<<<(16*H_)/256, 256, 0, stream>>>(db_w2, dbw2_t, H_, NH_, 16);     // [16,256] pad

  // 2) LN1
  ln_kernel<<<BN_, H_, 0, stream>>>(hidden, hn_g, hn_b, h_bf);

  // 3-5) q (scaled 1/sqrt(HD)), k, v projections -> head layouts
  {
    dim3 g(H_/64, BN_/16);
    gemm_bf16<<<g, 32, 0, stream>>>(h_bf, Wq_t, bq, nullptr, nullptr, nullptr, q_h,
                                    H_, H_, 0, 1, 0.17677669529663687f);
    gemm_bf16<<<g, 32, 0, stream>>>(h_bf, Wk_t, bk, nullptr, nullptr, nullptr, k_h,
                                    H_, H_, 0, 1, 1.f);
    gemm_bf16<<<g, 32, 0, stream>>>(h_bf, Wv_t, bv, nullptr, nullptr, nullptr, vT,
                                    H_, H_, 0, 2, 1.f);
  }

  // 6) distance-bias MLP (WMMA), 7) logits, 8) softmax, 9) head mean, 10) attn@v
  db_kernel<<<(B_*N_*N_)/16, 32, 0, stream>>>(coords, db_w1, db_b1, dbw2_t, db_b2, db);
  attn_logits_kernel<<<dim3(N_/16, B_*NH_*N_/16), 32, 0, stream>>>(q_h, k_h, db, mask, attn);
  softmax_kernel<<<B_*NH_*N_, 32, 0, stream>>>(attn, attnb, mask);
  attn_mean_kernel<<<(B_*N_*N_)/256, 256, 0, stream>>>(attn, am);
  attn_msg_kernel<<<B_*NH_*N_/16, 32, 0, stream>>>(attnb, vT, msgb);

  // 11) output proj + residual -> hid2
  gemm_bf16<<<dim3(H_/64, BN_/16), 32, 0, stream>>>(msgb, Wo_t, bo, hidden, nullptr,
                                                    hid2, nullptr, H_, H_, 0, 0, 1.f);
  // 12) LN2
  ln_kernel<<<BN_, H_, 0, stream>>>(hid2, ffn_g, ffn_b, hff);
  // 13) FFN up (silu)
  gemm_bf16<<<dim3(FF_/64, BN_/16), 32, 0, stream>>>(hff, ffw1_t, ff_b1, nullptr, nullptr,
                                                     nullptr, a1, H_, FF_, 1, 0, 1.f);
  // 14) FFN down + residual + mask -> d_out hidden (+ bf16 copy for gate path)
  gemm_bf16<<<dim3(H_/64, BN_/16), 32, 0, stream>>>(a1, ffw2_t, ff_b2, hid2, mask,
                                                    outHidden, h3_bf, FF_, H_, 0, 0, 1.f);
  // 15) coord-gate hidden MLP (silu)
  gemm_bf16<<<dim3(H_/64, BN_/16), 32, 0, stream>>>(h3_bf, cg1_t, cg_b1, nullptr, nullptr,
                                                    nullptr, g1, H_, H_, 1, 0, 1.f);
  // 16) gate scalar, 17) coords update -> d_out coords
  gate_kernel<<<BN_, 32, 0, stream>>>(g1, cg_w2, cg_b2, gate);
  coord_kernel<<<BN_, 32, 0, stream>>>(am, coords, gate, mask, outCoords);
}
